// SA_56444460204407
// MI455X (gfx1250) — matrix operand order repaired and checked by reading, not compiled, not run on hardware
//
#include <hip/hip_runtime.h>
#include <math.h>

// ---------------------------------------------------------------------------
// Fused ReLU-QKV-proj -> H-axis attention -> ReLU out-proj for MI455X (gfx1250)
// One workgroup per (b, w) pixel-column: 1024 WGs, 256 threads (8 x wave32).
// All intermediates live in 256 KB of LDS; WMMA bf16 16x16x32 everywhere.
// ---------------------------------------------------------------------------

typedef __bf16 bf16;
typedef __attribute__((ext_vector_type(16))) __bf16 v16bf;
typedef __attribute__((ext_vector_type(8)))  __bf16 v8bf;
typedef __attribute__((ext_vector_type(8)))  float  v8f;

#define BN 16
#define CC 128
#define HH 256
#define WW 64

#define LDS_BYTES (4 * 64 * 1024)

#define V8F_ZERO {0.f, 0.f, 0.f, 0.f, 0.f, 0.f, 0.f, 0.f}

static __device__ __forceinline__ v8f wmma_bf16(v16bf a, v16bf b, v8f c) {
  // (neg_a, A, neg_b, B, c_mod, C, reuse_a, reuse_b)
  return __builtin_amdgcn_wmma_f32_16x16x32_bf16(false, a, false, b,
                                                 (short)0, c, false, false);
}

// A-operand (16x32, 16-bit): lane m<16 holds row m, K = k0..k0+15;
// lane m+16 holds row m, K = k0+16..k0+31. One contiguous 32B LDS read per lane.
static __device__ __forceinline__ v16bf load_a16(const bf16* __restrict__ base,
                                                 int rs, int row0, int k0,
                                                 int lane) {
  const int row = row0 + (lane & 15);
  const int kb  = k0 + ((lane >> 4) << 4);   // +0 or +16
  const bf16* p = base + row * rs + kb;
  v16bf a;
#pragma unroll
  for (int i = 0; i < 16; ++i) a[i] = p[i];
  return a;
}

// B-operand (32x16, 16-bit): lane n<16 -> col n, K = k0..k0+15;
// lane n+16 -> col n, K = k0+16..k0+31. One contiguous 32B LDS read per lane.
static __device__ __forceinline__ v16bf load_b16(const bf16* __restrict__ base,
                                                 int rs, int k0, int col0,
                                                 int lane) {
  const int row = col0 + (lane & 15);
  const int kb  = k0 + ((lane >> 4) << 4);   // +0 or +16
  const bf16* p = base + row * rs + kb;
  v16bf b;
#pragma unroll
  for (int i = 0; i < 16; ++i) b[i] = p[i];
  return b;
}

// A-operand [o x c] from a global f32 row-major weight matrix (cvt to bf16).
static __device__ __forceinline__ v16bf load_a_w(const float* __restrict__ wmat,
                                                 int ldw, int o0, int c0,
                                                 int lane) {
  const int row = o0 + (lane & 15);
  const int kb  = c0 + ((lane >> 4) << 4);
  const float* p = wmat + row * ldw + kb;
  v16bf a;
#pragma unroll
  for (int i = 0; i < 16; ++i) a[i] = (bf16)p[i];
  return a;
}

// B-operand [c x o] (i.e. W^T) from a global f32 row-major weight matrix.
static __device__ __forceinline__ v16bf load_b_w(const float* __restrict__ wmat,
                                                 int ldw, int k0, int o0,
                                                 int lane) {
  const int row = o0 + (lane & 15);
  const int kb  = k0 + ((lane >> 4) << 4);
  const float* p = wmat + row * ldw + kb;
  v16bf b;
#pragma unroll
  for (int i = 0; i < 16; ++i) b[i] = (bf16)p[i];
  return b;
}

// relu(acc + bias[r]) -> packed bf16 x8
static __device__ __forceinline__ v8bf relu_pack(v8f acc, const float* bl) {
  v8bf pk;
#pragma unroll
  for (int r = 0; r < 8; ++r) {
    float y = acc[r] + bl[r];
    pk[r] = (bf16)(y > 0.f ? y : 0.f);
  }
  return pk;
}

// ---- Q/K projection: dst[h][c] = relu(W X + b), one o-strip per wave -------
static __device__ __forceinline__ void proj_qk(const bf16* __restrict__ Xt,
                                               bf16* __restrict__ dst,
                                               const float* __restrict__ wmat,
                                               const float* __restrict__ bias,
                                               int wave, int lane) {
  const int o0 = wave << 4;
  v16bf aw[4];
#pragma unroll
  for (int kc = 0; kc < 4; ++kc) aw[kc] = load_a_w(wmat, CC, o0, kc * 32, lane);
  float bl[8];
#pragma unroll
  for (int r = 0; r < 8; ++r) bl[r] = bias[o0 + ((lane >> 4) << 3) + r];

  const int hc    = lane & 15;
  const int obase = o0 + ((lane >> 4) << 3);
#pragma unroll 1
  for (int h0 = 0; h0 < HH; h0 += 32) {   // two tiles -> two indep. chains
    v16bf bx0[4], bx1[4];
#pragma unroll
    for (int kc = 0; kc < 4; ++kc) bx0[kc] = load_b16(Xt, CC, kc * 32, h0, lane);
#pragma unroll
    for (int kc = 0; kc < 4; ++kc) bx1[kc] = load_b16(Xt, CC, kc * 32, h0 + 16, lane);
    v8f a0 = V8F_ZERO, a1 = V8F_ZERO;
#pragma unroll
    for (int kc = 0; kc < 4; ++kc) {
      a0 = wmma_bf16(aw[kc], bx0[kc], a0);
      a1 = wmma_bf16(aw[kc], bx1[kc], a1);
    }
    *(v8bf*)(dst + (h0 + hc) * CC + obase)      = relu_pack(a0, bl);
    *(v8bf*)(dst + (h0 + 16 + hc) * CC + obase) = relu_pack(a1, bl);
  }
}

__global__ __launch_bounds__(256, 1)
void SA_fused_attn_kernel(const float* __restrict__ rep,
                          const float* __restrict__ wq, const float* __restrict__ bq,
                          const float* __restrict__ wk, const float* __restrict__ bk,
                          const float* __restrict__ wv, const float* __restrict__ bv,
                          const float* __restrict__ wfc, const float* __restrict__ bfc,
                          float* __restrict__ out) {
  extern __shared__ char smem[];
  bf16* Xt   = (bf16*)(smem);                 // [HH][CC]  X^T, bf16 (64 KB)
  bf16* Qs   = (bf16*)(smem + 1 * 65536);     // [HH][CC]  Q (row=h)
  bf16* Ks   = (bf16*)(smem + 2 * 65536);     // [HH][CC]  K (row=g)
  bf16* Vt   = (bf16*)(smem + 3 * 65536);     // [CC][HH]  V^T (row=c)
  bf16* Outs = Xt;                            // O^T reuses Xt after phase 1

  const int tid  = threadIdx.x;
  const int lane = tid & 31;
  const int wave = tid >> 5;
  const int b    = blockIdx.x >> 6;           // / WW
  const int w    = blockIdx.x & (WW - 1);

  // ---- Phase 0: stage X^T[h][c] = rep[b,c,h,w] into LDS as bf16 ----------
  const float* repb = rep + (size_t)b * CC * HH * WW + w;
#pragma unroll 4
  for (int idx = tid; idx < CC * HH; idx += 256) {
    const int c = idx >> 8;                   // 0..127
    const int h = idx & 255;                  // fastest: global stride WW floats
    Xt[h * CC + c] = (bf16)repb[(c * HH + h) * WW];
  }
  __syncthreads();

  // ---- Phase 1: Q,K = relu(W X + b) as [h][c]; V computed transposed -----
  proj_qk(Xt, Qs, wq, bq, wave, lane);
  proj_qk(Xt, Ks, wk, bk, wave, lane);
  {
    // V^T[c][h]: D' = (X^T rows h) x (Wv^T cols o) -> lane = col o, rows = h,
    // so each lane writes 8 consecutive h -> single 16B LDS store, and the
    // bias is a single per-lane scalar.
    const int o0 = wave << 4;
    v16bf bw[4];
#pragma unroll
    for (int kc = 0; kc < 4; ++kc) bw[kc] = load_b_w(wv, CC, kc * 32, o0, lane);
    const float bvl  = bv[o0 + (lane & 15)];
    const int   orow = o0 + (lane & 15);
    const int   hh8  = (lane >> 4) << 3;
#pragma unroll 1
    for (int h0 = 0; h0 < HH; h0 += 32) {
      v16bf ax0[4], ax1[4];
#pragma unroll
      for (int kc = 0; kc < 4; ++kc) ax0[kc] = load_a16(Xt, CC, h0, kc * 32, lane);
#pragma unroll
      for (int kc = 0; kc < 4; ++kc) ax1[kc] = load_a16(Xt, CC, h0 + 16, kc * 32, lane);
      v8f a0 = V8F_ZERO, a1 = V8F_ZERO;
#pragma unroll
      for (int kc = 0; kc < 4; ++kc) {
        a0 = wmma_bf16(ax0[kc], bw[kc], a0);
        a1 = wmma_bf16(ax1[kc], bw[kc], a1);
      }
      v8bf p0, p1;
#pragma unroll
      for (int r = 0; r < 8; ++r) {
        float y0 = a0[r] + bvl, y1 = a1[r] + bvl;
        p0[r] = (bf16)(y0 > 0.f ? y0 : 0.f);
        p1[r] = (bf16)(y1 > 0.f ? y1 : 0.f);
      }
      *(v8bf*)(Vt + orow * HH + h0 + hh8)      = p0;
      *(v8bf*)(Vt + orow * HH + h0 + 16 + hh8) = p1;
    }
  }
  __syncthreads();

  // ---- Phase 2: flash attention over H, computed transposed --------------
  // S^T = K * Q^T -> query index lands on the lane axis, so softmax state
  // (m, l) are per-lane scalars + one shfl_xor(16) to merge half-waves.
  const float SCALE = 0.08838834764831845f;   // 1/sqrt(128)
#pragma unroll 1
  for (int t = 0; t < 2; ++t) {
    const int h0 = ((wave << 1) + t) << 4;    // each wave: 2 query tiles of 16
    v16bf qb[4];                              // Q^T B-operand, kept resident
#pragma unroll
    for (int kc = 0; kc < 4; ++kc) qb[kc] = load_b16(Qs, CC, kc * 32, h0, lane);

    v8f oa[8];                                // O^T accum: [c x h], 64 VGPRs
#pragma unroll
    for (int ct = 0; ct < 8; ++ct) oa[ct] = (v8f)V8F_ZERO;
    float m_run = -1e30f, l_run = 0.f;

#pragma unroll 1
    for (int g0 = 0; g0 < HH; g0 += 32) {
      v8f sA = V8F_ZERO, sB = V8F_ZERO;
      {
        v16bf akA[4], akB[4];
#pragma unroll
        for (int kc = 0; kc < 4; ++kc) akA[kc] = load_a16(Ks, CC, g0, kc * 32, lane);
#pragma unroll
        for (int kc = 0; kc < 4; ++kc) akB[kc] = load_a16(Ks, CC, g0 + 16, kc * 32, lane);
#pragma unroll
        for (int kc = 0; kc < 4; ++kc) {
          sA = wmma_bf16(akA[kc], qb[kc], sA);
          sB = wmma_bf16(akB[kc], qb[kc], sB);
        }
      }
      // online softmax: lane holds 16 of the 32 g-scores for its query column
      float mx = m_run;
#pragma unroll
      for (int r = 0; r < 8; ++r) {
        sA[r] *= SCALE; sB[r] *= SCALE;
        mx = fmaxf(mx, fmaxf(sA[r], sB[r]));
      }
      mx = fmaxf(mx, __shfl_xor(mx, 16, 32));
      const float cf = __expf(m_run - mx);
      float pA[8], pB[8], rs = 0.f;
#pragma unroll
      for (int r = 0; r < 8; ++r) {
        pA[r] = __expf(sA[r] - mx);
        pB[r] = __expf(sB[r] - mx);
        rs += pA[r] + pB[r];
      }
      rs += __shfl_xor(rs, 16, 32);
      l_run = l_run * cf + rs;
      m_run = mx;
#pragma unroll
      for (int ct = 0; ct < 8; ++ct)
#pragma unroll
        for (int r = 0; r < 8; ++r) oa[ct][r] *= cf;

      // Rebuild P^T as a B-operand (32x16): half-wave exchange fills the
      // missing 8 g-values per lane.
      v16bf pb;
#pragma unroll
      for (int r = 0; r < 8; ++r) {
        const float xa = __shfl_xor(pA[r], 16, 32);
        const float xb = __shfl_xor(pB[r], 16, 32);
        pb[r]     = (bf16)((lane < 16) ? pA[r] : xb);
        pb[8 + r] = (bf16)((lane < 16) ? xa    : pB[r]);
      }
      // O^T += V^T(c x g) * P^T(g x h): batch loads, then 4-wmma bursts
      {
        v16bf av[4];
#pragma unroll
        for (int ct = 0; ct < 4; ++ct) av[ct] = load_a16(Vt, HH, ct * 16, g0, lane);
#pragma unroll
        for (int ct = 0; ct < 4; ++ct) oa[ct] = wmma_bf16(av[ct], pb, oa[ct]);
#pragma unroll
        for (int ct = 0; ct < 4; ++ct) av[ct] = load_a16(Vt, HH, (ct + 4) * 16, g0, lane);
#pragma unroll
        for (int ct = 0; ct < 4; ++ct) oa[ct + 4] = wmma_bf16(av[ct], pb, oa[ct + 4]);
      }
    }
    // normalize, write O^T -> Outs[h][c] (aliases Xt; Xt dead since barrier 1)
    const float inv   = 1.f / l_run;
    const int   hcol  = h0 + (lane & 15);
    const int   half8 = (lane >> 4) << 3;
#pragma unroll
    for (int ct = 0; ct < 8; ++ct) {
      v8bf pk;
#pragma unroll
      for (int r = 0; r < 8; ++r) pk[r] = (bf16)(oa[ct][r] * inv);
      *(v8bf*)(Outs + hcol * CC + ct * 16 + half8) = pk;
    }
  }
  __syncthreads();

  // ---- Phase 3: out[b,o,h,w] = relu(Wfc * O + bfc) -----------------------
  {
    const int o0 = wave << 4;                 // one o-strip per wave
    v16bf aw[4];
#pragma unroll
    for (int kc = 0; kc < 4; ++kc) aw[kc] = load_a_w(wfc, CC, o0, kc * 32, lane);
    float bl[8];
#pragma unroll
    for (int r = 0; r < 8; ++r) bl[r] = bfc[o0 + ((lane >> 4) << 3) + r];

    float* outb = out + (size_t)b * CC * HH * WW + w;
    const int hc = lane & 15;
    const int ob = o0 + ((lane >> 4) << 3);
#pragma unroll 1
    for (int h0 = 0; h0 < HH; h0 += 32) {
      v16bf bx0[4], bx1[4];
#pragma unroll
      for (int kc = 0; kc < 4; ++kc) bx0[kc] = load_b16(Outs, CC, kc * 32, h0, lane);
#pragma unroll
      for (int kc = 0; kc < 4; ++kc) bx1[kc] = load_b16(Outs, CC, kc * 32, h0 + 16, lane);
      v8f a0 = V8F_ZERO, a1 = V8F_ZERO;
#pragma unroll
      for (int kc = 0; kc < 4; ++kc) {
        a0 = wmma_bf16(aw[kc], bx0[kc], a0);
        a1 = wmma_bf16(aw[kc], bx1[kc], a1);
      }
#pragma unroll
      for (int r = 0; r < 8; ++r) {
        float y0 = a0[r] + bl[r];
        float y1 = a1[r] + bl[r];
        outb[((ob + r) * HH + h0 + hc) * WW]      = (y0 > 0.f ? y0 : 0.f);
        outb[((ob + r) * HH + h0 + 16 + hc) * WW] = (y1 > 0.f ? y1 : 0.f);
      }
    }
  }
}

extern "C" void kernel_launch(void* const* d_in, const int* in_sizes, int n_in,
                              void* d_out, int out_size, void* d_ws, size_t ws_size,
                              hipStream_t stream) {
  (void)in_sizes; (void)n_in; (void)out_size; (void)d_ws; (void)ws_size;
  const float* rep = (const float*)d_in[0];
  const float* wq  = (const float*)d_in[1];
  const float* bq  = (const float*)d_in[2];
  const float* wk  = (const float*)d_in[3];
  const float* bk  = (const float*)d_in[4];
  const float* wv  = (const float*)d_in[5];
  const float* bv  = (const float*)d_in[6];
  const float* wfc = (const float*)d_in[7];
  const float* bfc = (const float*)d_in[8];
  float* out = (float*)d_out;

  // 256 KB dynamic LDS per workgroup (WGP holds 320 KB on CDNA5).
  (void)hipFuncSetAttribute((const void*)SA_fused_attn_kernel,
                            hipFuncAttributeMaxDynamicSharedMemorySize,
                            LDS_BYTES);
  dim3 grid(BN * WW);   // 1024 workgroups: one per (b, w)
  dim3 block(256);      // 8 x wave32
  SA_fused_attn_kernel<<<grid, block, LDS_BYTES, stream>>>(
      rep, wq, bq, wk, bk, wv, bv, wfc, bfc, out);
}
